// Prior_Attention_51153060495900
// MI455X (gfx1250) — matrix operand order repaired and checked by reading, not compiled, not run on hardware
//
#include <hip/hip_runtime.h>
#include <hip/hip_bf16.h>

typedef __bf16 bf16_t;
typedef __attribute__((ext_vector_type(8)))  bf16_t v8bf;
typedef __attribute__((ext_vector_type(16))) bf16_t v16bf;
typedef __attribute__((ext_vector_type(8)))  float  v8f;
typedef __attribute__((ext_vector_type(4)))  int    v4i;

#define DIMC  384
#define HEADS 8
#define CHH   48
#define NSP   16384
#define BATCH 4

#define AS1 __attribute__((address_space(1)))
#define AS3 __attribute__((address_space(3)))
typedef AS1 v4i* gv4i_ptr;   // global int4*
typedef AS3 v4i* lv4i_ptr;   // LDS int4*

#if defined(__gfx1250__) && __has_builtin(__builtin_amdgcn_global_load_async_to_lds_b128) && __has_builtin(__builtin_amdgcn_s_wait_asynccnt)
#define USE_ASYNC_LDS 1
#else
#define USE_ASYNC_LDS 0
#endif

static __device__ __forceinline__ v16bf join16(v8bf lo, v8bf hi) {
  return __builtin_shufflevector(lo, hi, 0,1,2,3,4,5,6,7,8,9,10,11,12,13,14,15);
}

static __device__ __forceinline__ void wait_async_then_barrier() {
#if USE_ASYNC_LDS
  __builtin_amdgcn_s_wait_asynccnt(0);
#endif
  __syncthreads();
}

// ---------------------------------------------------------------------------
// Fold BN into conv1x1: W'[o,c] = s[o]*W[o,c] (bf16), b'[o] = s*b + beta - mean*s
// ---------------------------------------------------------------------------
__global__ void fold_bn(const float* __restrict__ w, const float* __restrict__ b,
                        const float* __restrict__ g, const float* __restrict__ beta,
                        const float* __restrict__ mean, const float* __restrict__ var,
                        bf16_t* __restrict__ wOut, float* __restrict__ bOut) {
  int i = blockIdx.x * blockDim.x + threadIdx.x;
  if (i >= DIMC * DIMC) return;
  int o = i / DIMC;
  float s = g[o] * rsqrtf(var[o] + 1e-5f);
  wOut[i] = (bf16_t)(w[i] * s);
  if ((i % DIMC) == 0) bOut[o] = b[o] * s + beta[o] - mean[o] * s;
}

// ---------------------------------------------------------------------------
// Tiled bf16 WMMA GEMM: D[b][m][n] = sum_k A[b][m][k] * Bsrc[b][k][n] (+bias[m])
// Block tile 64(M) x 128(N), 8 waves as 4(M)x2(N); wave = 16x64 (4 C-tiles).
// Double-buffered LDS, async global->LDS fills where the dtype allows.
// ---------------------------------------------------------------------------
template <typename TIN, typename TOUT>
__global__ __launch_bounds__(256)
void gemm_bf16_wmma(const bf16_t* __restrict__ A, long long aBatchStride,
                    const TIN* __restrict__ Bsrc, long long bBatchStride,
                    const float* __restrict__ bias,
                    TOUT* __restrict__ D, long long dBatchStride, int K) {
  __shared__ bf16_t Alds[2][64][32];
  __shared__ bf16_t Blds[2][32][128];
  const int b = blockIdx.z;
  const int mBase = blockIdx.x * 64;
  const int nBase = blockIdx.y * 128;
  A    += (long long)b * aBatchStride;
  Bsrc += (long long)b * bBatchStride;
  D    += (long long)b * dBatchStride;

  const int tid   = threadIdx.x;
  const int lane  = tid & 31;
  const int wave  = tid >> 5;
  const int waveM = wave & 3;    // 16-row subtile
  const int waveN = wave >> 2;   // 64-col group
  const int half  = lane >> 4;
  const int mrow  = lane & 15;

  const int rA = tid >> 2, cA = (tid & 3) * 8;   // A tile: 64x32, 16B/thread
  const int rB = tid >> 3, cB = (tid & 7) * 16;  // B tile: 32x128, 16 elems/thread

  auto loadTiles = [&](int buf, int k0) {
    // ---- A tile (always bf16) ----
    const bf16_t* aSrc = &A[(long long)(mBase + rA) * K + k0 + cA];
#if USE_ASYNC_LDS
    __builtin_amdgcn_global_load_async_to_lds_b128(
        (gv4i_ptr)aSrc, (lv4i_ptr)&Alds[buf][rA][cA], 0, 0);
#else
    *(uint4*)&Alds[buf][rA][cA] = *(const uint4*)aSrc;
#endif
    // ---- B tile ----
    const TIN* src = &Bsrc[(long long)(k0 + rB) * NSP + nBase + cB];
    if constexpr (sizeof(TIN) == 4) {
      __builtin_prefetch(src + 32LL * NSP, 0, 0);   // next tile, same lane slot
      const float4* s4 = (const float4*)src;
#pragma unroll
      for (int j = 0; j < 4; ++j) {
        float4 f = s4[j];
        Blds[buf][rB][cB + 4 * j + 0] = (bf16_t)f.x;
        Blds[buf][rB][cB + 4 * j + 1] = (bf16_t)f.y;
        Blds[buf][rB][cB + 4 * j + 2] = (bf16_t)f.z;
        Blds[buf][rB][cB + 4 * j + 3] = (bf16_t)f.w;
      }
    } else {
#if USE_ASYNC_LDS
      // INST_OFFSET applies to both global and LDS addresses (ISA 10.x async)
      __builtin_amdgcn_global_load_async_to_lds_b128(
          (gv4i_ptr)src, (lv4i_ptr)&Blds[buf][rB][cB], 0, 0);
      __builtin_amdgcn_global_load_async_to_lds_b128(
          (gv4i_ptr)src, (lv4i_ptr)&Blds[buf][rB][cB], 16, 0);
#else
      ((uint4*)&Blds[buf][rB][cB])[0] = ((const uint4*)src)[0];
      ((uint4*)&Blds[buf][rB][cB])[1] = ((const uint4*)src)[1];
#endif
    }
  };

  v8f acc[4] = {};
  loadTiles(0, 0);
  int buf = 0;

  for (int k0 = 0; k0 < K; k0 += 32) {
    wait_async_then_barrier();
    if (k0 + 32 < K) loadTiles(buf ^ 1, k0 + 32);

    // A fragment: lane row m, k-runs [8h..8h+7] and [16+8h..16+8h+7]
    v8bf alo = *(const v8bf*)&Alds[buf][waveM * 16 + mrow][half * 8];
    v8bf ahi = *(const v8bf*)&Alds[buf][waveM * 16 + mrow][16 + half * 8];
    v16bf afrag = join16(alo, ahi);

#pragma unroll
    for (int t = 0; t < 4; ++t) {
      int col  = waveN * 64 + t * 16;
      v16bf bfrag;   // lane = N col, elems over k in table order
#pragma unroll
      for (int s = 0; s < 16; ++s)
        bfrag[s] = Blds[buf][(s < 8) ? (half * 8 + s) : (16 + half * 8 + (s - 8))][col + mrow];
      acc[t] = __builtin_amdgcn_wmma_f32_16x16x32_bf16(
          false, afrag, false, bfrag, (short)0, acc[t], false, false);
    }
    buf ^= 1;
  }

  // epilogue: lane -> N = mrow, VGPR v -> M = v + 8*half
#pragma unroll
  for (int t = 0; t < 4; ++t) {
    int n = nBase + waveN * 64 + t * 16 + mrow;
#pragma unroll
    for (int v = 0; v < 8; ++v) {
      int m = mBase + waveM * 16 + v + 8 * half;
      float val = acc[t][v];
      if (bias) val += bias[m];
      D[(long long)m * NSP + n] = (TOUT)val;
    }
  }
}

// ---------------------------------------------------------------------------
// Depthwise 3x3, SAME padding; bf16 in/out, f32 weights/accum.
// ---------------------------------------------------------------------------
__global__ __launch_bounds__(256)
void dwconv3x3(const bf16_t* __restrict__ in, const float* __restrict__ w,
               bf16_t* __restrict__ out, long long outBatchStride) {
  long long i = (long long)blockIdx.x * 256 + threadIdx.x;
  int x = (int)(i & 127);
  int y = (int)((i >> 7) & 127);
  int c = (int)((i >> 14) % DIMC);
  int b = (int)(i / ((long long)DIMC * NSP));
  const bf16_t* src = in + (long long)(b * DIMC + c) * NSP;
  const float* wc = w + c * 9;
  float acc = 0.f;
#pragma unroll
  for (int dy = -1; dy <= 1; ++dy)
#pragma unroll
    for (int dx = -1; dx <= 1; ++dx) {
      int yy = y + dy, xx = x + dx;
      if (yy >= 0 && yy < 128 && xx >= 0 && xx < 128)
        acc += (float)src[yy * 128 + xx] * wc[(dy + 1) * 3 + (dx + 1)];
    }
  out[(long long)b * outBatchStride + (long long)c * NSP + y * 128 + x] = (bf16_t)acc;
}

// ---------------------------------------------------------------------------
// Per-(b,h): S = q k^T over K=16384 via WMMA (9 waves = 3x3 16-tiles of 48x48),
// k transposed into LDS at store time; B-fragments are gathered per element down a ksT column.
// Then row softmax(S * temperature[h]) -> attn f32 [48][48].
// ---------------------------------------------------------------------------
__global__ __launch_bounds__(288)
void attn_qk_softmax(const bf16_t* __restrict__ Q, const bf16_t* __restrict__ Kt,
                     const float* __restrict__ temperature,
                     float* __restrict__ attnOut) {
  __shared__ bf16_t qs[2][48][32];
  __shared__ bf16_t ksT[2][32][56];   // [n][d], row pad 56 (112B, 16B aligned)
  __shared__ float S[48][49];
  const int bh = blockIdx.x;           // b*8 + h
  const int h  = bh & 7;
  const bf16_t* qb = Q  + (long long)((bh >> 3) * DIMC + h * CHH) * NSP;
  const bf16_t* kb = Kt + (long long)((bh >> 3) * DIMC + h * CHH) * NSP;
  const int tid = threadIdx.x, lane = tid & 31, wave = tid >> 5;
  const int ti = wave / 3, tj = wave % 3;
  const int half = lane >> 4, mrow = lane & 15;

  auto loadQK = [&](int buf, int k0) {
    if (tid < 192) {                 // waves 0..5: uniform per-wave predicate
      int r = tid >> 2;              // channel 0..47
      int cOf = (tid & 3) * 8;       // spatial chunk
      const bf16_t* qp = &qb[(long long)r * NSP + k0 + cOf];
#if USE_ASYNC_LDS
      __builtin_amdgcn_global_load_async_to_lds_b128(
          (gv4i_ptr)qp, (lv4i_ptr)&qs[buf][r][cOf], 0, 0);
#else
      *(uint4*)&qs[buf][r][cOf] = *(const uint4*)qp;
#endif
      bf16_t tmp[8];
      *(uint4*)tmp = *(const uint4*)&kb[(long long)r * NSP + k0 + cOf];
#pragma unroll
      for (int j = 0; j < 8; ++j) ksT[buf][cOf + j][r] = tmp[j];  // transpose
    }
  };

  v8f acc = {};
  loadQK(0, 0);
  int buf = 0;

  for (int k0 = 0; k0 < NSP; k0 += 32) {
    wait_async_then_barrier();
    if (k0 + 32 < NSP) loadQK(buf ^ 1, k0 + 32);

    v8bf alo = *(const v8bf*)&qs[buf][ti * 16 + mrow][half * 8];
    v8bf ahi = *(const v8bf*)&qs[buf][ti * 16 + mrow][16 + half * 8];
    v16bf afrag = join16(alo, ahi);
    // B[kk=k(s)][nn=mrow] = ksT[k(s)][16tj+mrow], k(s) in table order.
    v16bf bfrag;
#pragma unroll
    for (int s = 0; s < 16; ++s)
      bfrag[s] = ksT[buf][(s < 8) ? (half * 8 + s) : (16 + half * 8 + (s - 8))][tj * 16 + mrow];
    acc = __builtin_amdgcn_wmma_f32_16x16x32_bf16(
        false, afrag, false, bfrag, (short)0, acc, false, false);
    buf ^= 1;
  }

#pragma unroll
  for (int v = 0; v < 8; ++v)
    S[ti * 16 + v + 8 * half][tj * 16 + mrow] = acc[v];
  __syncthreads();

  if (tid < 48) {
    float temp = temperature[h];
    float row[48];
    float mx = -1e30f;
    for (int d = 0; d < 48; ++d) { row[d] = S[tid][d] * temp; mx = fmaxf(mx, row[d]); }
    float sum = 0.f;
    for (int d = 0; d < 48; ++d) { row[d] = __expf(row[d] - mx); sum += row[d]; }
    float inv = 1.f / sum;
    float* o = attnOut + ((long long)bh * 48 + tid) * 48;
    for (int d = 0; d < 48; ++d) o[d] = row[d] * inv;
  }
}

// ---------------------------------------------------------------------------
// M[b][o][g] (g<384: proj_x * A_blk; g>=384: proj_p * A_blk), bf16 out.
// ---------------------------------------------------------------------------
__global__ __launch_bounds__(256)
void build_m(const float* __restrict__ proj, const float* __restrict__ attn,
             bf16_t* __restrict__ Mout) {
  int i = blockIdx.x * 256 + threadIdx.x;
  if (i >= BATCH * DIMC * 768) return;
  int g = i % 768;
  int o = (i / 768) % DIMC;
  int b = i / (768 * DIMC);
  int sel = g / DIMC, gp = g % DIMC;
  int h = gp / CHH, d = gp % CHH;
  const float* pr = proj + (long long)o * 768 + sel * DIMC + h * CHH;
  const float* at = attn + ((long long)(b * HEADS + h) * CHH) * CHH + d;
  float s = 0.f;
#pragma unroll 8
  for (int cp = 0; cp < CHH; ++cp) s += pr[cp] * at[cp * CHH];
  Mout[i] = (bf16_t)s;
}

// ---------------------------------------------------------------------------
extern "C" void kernel_launch(void* const* d_in, const int* in_sizes, int n_in,
                              void* d_out, int out_size, void* d_ws, size_t ws_size,
                              hipStream_t stream) {
  const float* x    = (const float*)d_in[0];
  const float* p    = (const float*)d_in[1];
  const float* temp = (const float*)d_in[2];
  const float* proj = (const float*)d_in[31];

  char* ws = (char*)d_ws;
  const size_t SZ_WF = (size_t)4 * DIMC * DIMC * 2;
  const size_t SZ_BF = (size_t)4 * DIMC * 4;
  const size_t SZ_T  = (size_t)BATCH * DIMC * NSP * 2;   // bf16 tensor
  bf16_t* Wf = (bf16_t*)(ws);
  float*  Bf = (float*)(ws + SZ_WF);
  bf16_t* Yb = (bf16_t*)(ws + SZ_WF + SZ_BF);
  bf16_t* Qb = (bf16_t*)(ws + SZ_WF + SZ_BF + SZ_T);
  bf16_t* Kb = (bf16_t*)(ws + SZ_WF + SZ_BF + 2 * SZ_T);
  bf16_t* Vb = (bf16_t*)(ws + SZ_WF + SZ_BF + 3 * SZ_T);          // [B][768][NSP]
  float*  At = (float*)(ws + SZ_WF + SZ_BF + 5 * SZ_T);
  bf16_t* Mb = (bf16_t*)(ws + SZ_WF + SZ_BF + 5 * SZ_T + (size_t)BATCH * HEADS * CHH * CHH * 4);

  // 1) Fold BN into the four conv1x1 weights.  conv idx: 0=k 1=v 2=pq 3=pv
  const int wi[4] = {3, 5, 7, 9};     // weight idx (bias = +1)
  const int gi[4] = {11, 15, 19, 23}; // g, beta, mean, var consecutive
  dim3 fgrid((DIMC * DIMC + 255) / 256);
  for (int c = 0; c < 4; ++c)
    fold_bn<<<fgrid, 256, 0, stream>>>(
        (const float*)d_in[wi[c]], (const float*)d_in[wi[c] + 1],
        (const float*)d_in[gi[c]], (const float*)d_in[gi[c] + 1],
        (const float*)d_in[gi[c] + 2], (const float*)d_in[gi[c] + 3],
        Wf + (size_t)c * DIMC * DIMC, Bf + c * DIMC);

  dim3 ggrid(DIMC / 64, NSP / 128, BATCH);
  const long long xBS = (long long)DIMC * NSP;
  const long long vBS = (long long)768 * NSP;
  const int dwBlocks = (int)(((long long)BATCH * DIMC * NSP) / 256);

  // 2) conv k(x) -> Y; dwconv dwk -> Kb
  gemm_bf16_wmma<float, bf16_t><<<ggrid, 256, 0, stream>>>(
      Wf + 0 * DIMC * DIMC, 0, x, xBS, Bf + 0 * DIMC, Yb, xBS, DIMC);
  dwconv3x3<<<dwBlocks, 256, 0, stream>>>(Yb, (const float*)d_in[27], Kb, xBS);
  // 3) conv v(x) -> Y; dwconv dwv -> Vb[:,0:384]
  gemm_bf16_wmma<float, bf16_t><<<ggrid, 256, 0, stream>>>(
      Wf + 1 * DIMC * DIMC, 0, x, xBS, Bf + 1 * DIMC, Yb, xBS, DIMC);
  dwconv3x3<<<dwBlocks, 256, 0, stream>>>(Yb, (const float*)d_in[28], Vb, vBS);
  // 4) conv pq(p) -> Y; dwconv dwq -> Qb
  gemm_bf16_wmma<float, bf16_t><<<ggrid, 256, 0, stream>>>(
      Wf + 2 * DIMC * DIMC, 0, p, xBS, Bf + 2 * DIMC, Yb, xBS, DIMC);
  dwconv3x3<<<dwBlocks, 256, 0, stream>>>(Yb, (const float*)d_in[29], Qb, xBS);
  // 5) conv pv(p) -> Y; dwconv dwpv -> Vb[:,384:768]
  gemm_bf16_wmma<float, bf16_t><<<ggrid, 256, 0, stream>>>(
      Wf + 3 * DIMC * DIMC, 0, p, xBS, Bf + 3 * DIMC, Yb, xBS, DIMC);
  dwconv3x3<<<dwBlocks, 256, 0, stream>>>(Yb, (const float*)d_in[30],
                                          Vb + (size_t)DIMC * NSP, vBS);

  // 6) attention Gram + softmax per (b,h)
  attn_qk_softmax<<<BATCH * HEADS, 288, 0, stream>>>(Qb, Kb, temp, At);

  // 7) fold proj through block-diagonal attention: M[b] = [P1*A_blk | P2*A_blk]
  build_m<<<(BATCH * DIMC * 768 + 255) / 256, 256, 0, stream>>>(proj, At, Mb);

  // 8) final fused GEMM: out[b] = M[b] @ concat(vx, vp)[b]   (K = 768), f32 out
  gemm_bf16_wmma<bf16_t, float><<<ggrid, 256, 0, stream>>>(
      Mb, (long long)DIMC * 768, Vb, vBS, nullptr,
      (float*)d_out, xBS, 768);
}
